// SensorGCN_82575041232963
// MI455X (gfx1250) — compile-verified, matrix operand order checked by reading, not run on hardware
//
#include <hip/hip_runtime.h>
#include <hip/hip_bf16.h>

typedef __attribute__((ext_vector_type(16))) __bf16 v16bf;
typedef __attribute__((ext_vector_type(8)))  __bf16 v8bf;
typedef __attribute__((ext_vector_type(8)))  float  v8f;

#define H 64
#define NGRAPH 64

// ---------------------------------------------------------------- utilities
__global__ void fill_kernel(float* p, float v, int n) {
    int i = blockIdx.x * blockDim.x + threadIdx.x;
    if (i < n) p[i] = v;
}

// ---------------------------------------------------------------- degree / norm
__global__ void deg_edge_kernel(const int* __restrict__ ei,
                                const float* __restrict__ ew,
                                float* __restrict__ deg, int E) {
    int e = blockIdx.x * blockDim.x + threadIdx.x;
    if (e < E) {
        int d = ei[E + e];
        atomicAdd(&deg[d], ew[e]);
    }
}

__global__ void dinv_kernel(float* __restrict__ deg, int n) {
    int i = blockIdx.x * blockDim.x + threadIdx.x;
    if (i < n) {
        float d = deg[i];
        deg[i] = (d > 0.0f) ? rsqrtf(d) : 0.0f;   // in-place: deg -> dinv
    }
}

__global__ void edge_norm_kernel(const int* __restrict__ ei,
                                 const float* __restrict__ ew,
                                 const float* __restrict__ dinv,
                                 float* __restrict__ norm, int E) {
    int e = blockIdx.x * blockDim.x + threadIdx.x;
    if (e < E) {
        int s = ei[e], d = ei[E + e];
        norm[e] = dinv[s] * ew[e] * dinv[d];
    }
}

// ---------------------------------------------------------------- layer-1 GEMM (K=5, VALU)
__global__ void gemm_k5_kernel(const float* __restrict__ x,
                               const float* __restrict__ W,   // 5 x 64
                               float* __restrict__ out, int n) {
    int idx = blockIdx.x * blockDim.x + threadIdx.x;
    if (idx >= n * H) return;
    int node = idx >> 6, f = idx & 63;
    const float* xr = x + (size_t)node * 5;
    float acc = 0.0f;
#pragma unroll
    for (int k = 0; k < 5; ++k) acc = fmaf(xr[k], W[k * H + f], acc);
    out[idx] = acc;
}

// ---------------------------------------------------------------- W -> bf16 transposed copy
// Wt[n*64 + k] = (bf16) W[k*64 + n]  so a lane's output-column fragment is
// contiguous in K and loads as b128.
__global__ void wcvt_kernel(const float* __restrict__ W, __bf16* __restrict__ Wt) {
    int idx = blockIdx.x * blockDim.x + threadIdx.x;   // 4096
    if (idx >= H * H) return;
    int ncol = idx >> 6, k = idx & 63;
    Wt[idx] = (__bf16)W[k * H + ncol];
}

// ---------------------------------------------------------------- WMMA GEMM: (n x 64) @ (64 x 64)
// A (bf16, row-major) 16-bit A-matrix 16x32 striping (ISA 7.12.2): lanes 0-15
// hold K 0..7 / 16..23, lanes 16-31 hold K 8..15 / 24..31 for row M = lane&15.
// B from Wt (bf16, transposed): lane holds one output column, same K pattern.
// All fragment loads are 16B contiguous -> global_load_b128, zero converts.
__global__ void gemm_bf16_wmma_kernel(const __bf16* __restrict__ Abf,  // n x 64
                                      const __bf16* __restrict__ Wt,   // 64 x 64 (transposed)
                                      float* __restrict__ Out,         // n x 64
                                      int n) {
    int wave = threadIdx.x >> 5;                 // 4 waves / block
    int row0 = (blockIdx.x * 4 + wave) * 16;
    if (row0 >= n) return;
    int lane = threadIdx.x & 31;
    int m    = lane & 15;
    int kb   = (lane >> 4) * 8;                  // 0 or 8

    const __bf16* arow = Abf + (size_t)(row0 + m) * H;
    v8bf al0 = *(const v8bf*)(arow + kb);
    v8bf ah0 = *(const v8bf*)(arow + kb + 16);
    v8bf al1 = *(const v8bf*)(arow + 32 + kb);
    v8bf ah1 = *(const v8bf*)(arow + 32 + kb + 16);
    v16bf a0 = __builtin_shufflevector(al0, ah0, 0,1,2,3,4,5,6,7,8,9,10,11,12,13,14,15);
    v16bf a1 = __builtin_shufflevector(al1, ah1, 0,1,2,3,4,5,6,7,8,9,10,11,12,13,14,15);

#pragma unroll
    for (int nt = 0; nt < 4; ++nt) {
        int ncol = nt * 16 + (lane & 15);
        const __bf16* wrow = Wt + (size_t)ncol * H;
        v8bf bl0 = *(const v8bf*)(wrow + kb);
        v8bf bh0 = *(const v8bf*)(wrow + kb + 16);
        v8bf bl1 = *(const v8bf*)(wrow + 32 + kb);
        v8bf bh1 = *(const v8bf*)(wrow + 32 + kb + 16);
        v16bf b0 = __builtin_shufflevector(bl0, bh0, 0,1,2,3,4,5,6,7,8,9,10,11,12,13,14,15);
        v16bf b1 = __builtin_shufflevector(bl1, bh1, 0,1,2,3,4,5,6,7,8,9,10,11,12,13,14,15);

        v8f c = {};
        c = __builtin_amdgcn_wmma_f32_16x16x32_bf16(false, a0, false, b0,
                                                    (short)0, c, false, false);
        c = __builtin_amdgcn_wmma_f32_16x16x32_bf16(false, a1, false, b1,
                                                    (short)0, c, false, false);
        int mbase = (lane < 16) ? 0 : 8;
#pragma unroll
        for (int r = 0; r < 8; ++r)
            Out[(size_t)(row0 + mbase + r) * H + ncol] = c[r];
    }
}

// ---------------------------------------------------------------- self-loop + bias init
__global__ void selfloop_bias_kernel(const float* __restrict__ xw,
                                     const float* __restrict__ dinv,
                                     const float* __restrict__ bias,
                                     float* __restrict__ out, int n) {
    int idx = blockIdx.x * blockDim.x + threadIdx.x;
    if (idx >= n * H) return;
    int node = idx >> 6, f = idx & 63;
    float di = dinv[node];
    out[idx] = xw[idx] * di * di + bias[f];
}

// ---------------------------------------------------------------- edge scatter-add
__global__ void scatter_kernel(const int* __restrict__ ei,
                               const float* __restrict__ norm,
                               const float* __restrict__ xw,
                               float* __restrict__ out, int E) {
    long long idx = (long long)blockIdx.x * blockDim.x + threadIdx.x;
    if (idx >= (long long)E * 16) return;
    int e = (int)(idx >> 4);
    int g = ((int)idx & 15) * 4;
    int s = ei[e], d = ei[E + e];
    float w = norm[e];
    const float4 v = *(const float4*)(xw + (size_t)s * H + g);
    float* o = out + (size_t)d * H + g;
    atomicAdd(o + 0, v.x * w);
    atomicAdd(o + 1, v.y * w);
    atomicAdd(o + 2, v.z * w);
    atomicAdd(o + 3, v.w * w);
}

// ---------------------------------------------------------------- batchnorm stats (sum, sumsq)
__global__ void bn_stats_kernel(const float* __restrict__ h,
                                float* __restrict__ acc,   // [128]: sum, sumsq
                                int n) {
    __shared__ float ls[H], ls2[H];
    int f  = threadIdx.x & 63;
    int rg = threadIdx.x >> 6;                 // 4 row-groups per 256-thread block
    float s = 0.0f, s2 = 0.0f;
    for (int node = blockIdx.x * 4 + rg; node < n; node += gridDim.x * 4) {
        float v = h[(size_t)node * H + f];
        s += v; s2 += v * v;
    }
    if (rg == 0) { ls[f] = 0.0f; ls2[f] = 0.0f; }
    __syncthreads();
    atomicAdd(&ls[f], s);
    atomicAdd(&ls2[f], s2);
    __syncthreads();
    if (rg == 0) {
        atomicAdd(&acc[f],      ls[f]);
        atomicAdd(&acc[f + H],  ls2[f]);
    }
}

// bn apply + optional relu; also emits a packed bf16 copy so the next layer's
// WMMA A-matrix loads are half-width and convert-free.
__global__ void bn_apply_kernel(float* __restrict__ h,
                                __bf16* __restrict__ hbf,
                                const float* __restrict__ acc,
                                const float* __restrict__ g,
                                const float* __restrict__ b,
                                int n, int relu) {
    int idx = blockIdx.x * blockDim.x + threadIdx.x;
    if (idx >= n * H) return;
    int f = idx & 63;
    float inv_n = 1.0f / (float)n;
    float mu  = acc[f] * inv_n;
    float var = acc[f + H] * inv_n - mu * mu;
    float sc  = rsqrtf(var + 1e-5f) * g[f];
    float v   = (h[idx] - mu) * sc + b[f];
    if (relu) v = fmaxf(v, 0.0f);
    h[idx]   = v;
    hbf[idx] = (__bf16)v;
}

// ---------------------------------------------------------------- pooling
__global__ void pool_kernel(const float* __restrict__ h,
                            const int* __restrict__ batch,
                            float* __restrict__ pooled, int n) {
    int idx = blockIdx.x * blockDim.x + threadIdx.x;
    if (idx >= n * H) return;
    int node = idx >> 6, f = idx & 63;
    atomicAdd(&pooled[batch[node] * H + f], h[idx]);
}

__global__ void count_kernel(const int* __restrict__ batch,
                             float* __restrict__ cnt, int n) {
    int i = blockIdx.x * blockDim.x + threadIdx.x;
    if (i < n) atomicAdd(&cnt[batch[i]], 1.0f);
}

// ---------------------------------------------------------------- MLP heads
__global__ void heads_kernel(const float* __restrict__ pooled,
                             const float* __restrict__ cnt,
                             const float* __restrict__ Wc1, const float* __restrict__ bc1,
                             const float* __restrict__ Wc2, const float* __restrict__ bc2,
                             const float* __restrict__ Wr1, const float* __restrict__ br1,
                             const float* __restrict__ Wr2, const float* __restrict__ br2,
                             float* __restrict__ out) {
    int g = threadIdx.x;
    if (g >= NGRAPH) return;
    float p[H];
    float c = fmaxf(cnt[g], 1.0f);
#pragma unroll
    for (int f = 0; f < H; ++f) p[f] = pooled[g * H + f] / c;

    // classification head: 64 -> 32 (relu) -> 3
    float hc[32];
    for (int j = 0; j < 32; ++j) {
        float a = bc1[j];
        for (int f = 0; f < H; ++f) a = fmaf(p[f], Wc1[f * 32 + j], a);
        hc[j] = fmaxf(a, 0.0f);
    }
    for (int k = 0; k < 3; ++k) {
        float a = bc2[k];
        for (int j = 0; j < 32; ++j) a = fmaf(hc[j], Wc2[j * 3 + k], a);
        out[g * 3 + k] = a;
    }

    // regression head: 64 -> 32 (relu) -> 1
    float hr[32];
    for (int j = 0; j < 32; ++j) {
        float a = br1[j];
        for (int f = 0; f < H; ++f) a = fmaf(p[f], Wr1[f * 32 + j], a);
        hr[j] = fmaxf(a, 0.0f);
    }
    float a = br2[0];
    for (int j = 0; j < 32; ++j) a = fmaf(hr[j], Wr2[j], a);
    out[NGRAPH * 3 + g] = a;
}

// ================================================================ launch
extern "C" void kernel_launch(void* const* d_in, const int* in_sizes, int n_in,
                              void* d_out, int out_size, void* d_ws, size_t ws_size,
                              hipStream_t stream) {
    const float* x     = (const float*)d_in[0];
    const int*   ei    = (const int*)  d_in[1];
    const float* ew    = (const float*)d_in[2];
    const int*   batch = (const int*)  d_in[3];
    const float* W1 = (const float*)d_in[4],  *b1 = (const float*)d_in[5];
    const float* W2 = (const float*)d_in[6],  *b2 = (const float*)d_in[7];
    const float* W3 = (const float*)d_in[8],  *b3 = (const float*)d_in[9];
    const float* bn_g[3] = {(const float*)d_in[10], (const float*)d_in[12], (const float*)d_in[14]};
    const float* bn_b[3] = {(const float*)d_in[11], (const float*)d_in[13], (const float*)d_in[15]};
    const float* Wc1 = (const float*)d_in[16], *bc1 = (const float*)d_in[17];
    const float* Wc2 = (const float*)d_in[18], *bc2 = (const float*)d_in[19];
    const float* Wr1 = (const float*)d_in[20], *br1 = (const float*)d_in[21];
    const float* Wr2 = (const float*)d_in[22], *br2 = (const float*)d_in[23];
    float* out = (float*)d_out;

    const int N = in_sizes[3];       // batch has N entries
    const int E = in_sizes[2];       // edge_attr has E entries

    // workspace layout (16B-aligned sections)
    float*  dinv   = (float*)d_ws;                       // N  (deg -> dinv)
    float*  norm   = dinv + ((N + 3) & ~3);              // E
    float*  P      = norm + ((E + 3) & ~3);              // N*H  feature buffer (f32)
    float*  Q      = P + (size_t)N * H;                  // N*H  gemm output
    float*  bnacc  = Q + (size_t)N * H;                  // 128
    float*  pooled = bnacc + 2 * H;                      // 64*64
    float*  cnt    = pooled + NGRAPH * H;                // 64
    __bf16* Pbf    = (__bf16*)(cnt + NGRAPH);            // N*H  bf16 activations
    __bf16* Wtbf   = Pbf + (size_t)N * H;                // 64*64 bf16 transposed W

    const int TB = 256;
    const int nh_blocks = (N * H + TB - 1) / TB;
    const int e_blocks  = (E + TB - 1) / TB;
    const int n_blocks  = (N + TB - 1) / TB;
    const long long sc_threads = (long long)E * 16;
    const int sc_blocks = (int)((sc_threads + TB - 1) / TB);
    const int gemm_blocks = (((N + 15) / 16) + 3) / 4;

    // ---- degree / symmetric norm (shared across all 3 layers)
    fill_kernel<<<n_blocks, TB, 0, stream>>>(dinv, 1.0f, N);          // self-loop weight
    deg_edge_kernel<<<e_blocks, TB, 0, stream>>>(ei, ew, dinv, E);
    dinv_kernel<<<n_blocks, TB, 0, stream>>>(dinv, N);
    edge_norm_kernel<<<e_blocks, TB, 0, stream>>>(ei, ew, dinv, norm, E);

    const float* lin_W[3] = {W1, W2, W3};
    const float* lin_b[3] = {b1, b2, b3};

    for (int layer = 0; layer < 3; ++layer) {
        // GEMM: features @ W -> Q
        if (layer == 0) {
            gemm_k5_kernel<<<nh_blocks, TB, 0, stream>>>(x, lin_W[0], Q, N);
        } else {
            wcvt_kernel<<<(H * H + TB - 1) / TB, TB, 0, stream>>>(lin_W[layer], Wtbf);
            gemm_bf16_wmma_kernel<<<gemm_blocks, 128, 0, stream>>>(Pbf, Wtbf, Q, N);
        }

        // aggregate: self-loop + bias init, then edge scatter-add -> P
        selfloop_bias_kernel<<<nh_blocks, TB, 0, stream>>>(Q, dinv, lin_b[layer], P, N);
        scatter_kernel<<<sc_blocks, TB, 0, stream>>>(ei, norm, Q, P, E);

        // batchnorm (+ relu on layers 0,1); also emits bf16 activations
        fill_kernel<<<1, 2 * H, 0, stream>>>(bnacc, 0.0f, 2 * H);
        bn_stats_kernel<<<1024, TB, 0, stream>>>(P, bnacc, N);
        bn_apply_kernel<<<nh_blocks, TB, 0, stream>>>(P, Pbf, bnacc,
                                                      bn_g[layer], bn_b[layer],
                                                      N, layer < 2 ? 1 : 0);
    }

    // global mean pool + heads
    fill_kernel<<<1, NGRAPH * H, 0, stream>>>(pooled, 0.0f, NGRAPH * H);
    fill_kernel<<<1, NGRAPH, 0, stream>>>(cnt, 0.0f, NGRAPH);
    pool_kernel<<<nh_blocks, TB, 0, stream>>>(P, batch, pooled, N);
    count_kernel<<<n_blocks, TB, 0, stream>>>(batch, cnt, N);
    heads_kernel<<<1, NGRAPH, 0, stream>>>(pooled, cnt, Wc1, bc1, Wc2, bc2,
                                           Wr1, br1, Wr2, br2, out);
    (void)n_in; (void)out_size; (void)ws_size;
}
